// GCNLayer_5514738008182
// MI455X (gfx1250) — matrix operand order repaired and checked by reading, not compiled, not run on hardware
//
#include <hip/hip_runtime.h>
#include <hip/hip_bf16.h>

// ---------------------------------------------------------------------------
// GCN layer for MI455X (gfx1250, wave32, WMMA).
//   out = (D^-1/2 (A+I) D^-1/2) @ X @ W^T + b
// Strategy: split-bf16 (hi+lo) emulated-fp32 GEMM on v_wmma_f32_16x16x32_bf16.
// B-panel staged with GLOBAL_LOAD_ASYNC_TO_LDS_B128 (ASYNCcnt), A-panel
// streamed with non-temporal loads (TH_NT) to protect L2-resident operands.
// ---------------------------------------------------------------------------

typedef __attribute__((ext_vector_type(16))) __bf16   v16bf;
typedef __attribute__((ext_vector_type(8)))  float    v8f;
typedef __attribute__((ext_vector_type(4)))  float    f32x4;
typedef __attribute__((ext_vector_type(4)))  unsigned u32x4;

union BFrag {
    v16bf v;
    u32x4 q[2];
};

#define NDIM 8192
#define FDIM 128

// Truncation split: hi = top 16 bits (exact residual), lo = RNE(residual).
__device__ __forceinline__ void split1(float x, unsigned short& h, unsigned short& l) {
    unsigned u = __builtin_bit_cast(unsigned, x);
    h = (unsigned short)(u >> 16);
    float fh = __builtin_bit_cast(float, u & 0xFFFF0000u);
    float r  = x - fh;
    unsigned ur = __builtin_bit_cast(unsigned, r);
    l = (unsigned short)((ur + 0x7FFFu + ((ur >> 16) & 1u)) >> 16);
}

__device__ __forceinline__ void split_store4(unsigned short* ph, unsigned short* pl, f32x4 v) {
    unsigned short h0, h1, h2, h3, l0, l1, l2, l3;
    split1(v[0], h0, l0); split1(v[1], h1, l1);
    split1(v[2], h2, l2); split1(v[3], h3, l3);
    unsigned long long H = (unsigned long long)h0 | ((unsigned long long)h1 << 16)
                         | ((unsigned long long)h2 << 32) | ((unsigned long long)h3 << 48);
    unsigned long long L = (unsigned long long)l0 | ((unsigned long long)l1 << 16)
                         | ((unsigned long long)l2 << 32) | ((unsigned long long)l3 << 48);
    *(unsigned long long*)ph = H;
    *(unsigned long long*)pl = L;
}

__device__ __forceinline__ v8f wmma_bf16(const BFrag& a, const BFrag& b, v8f c) {
    return __builtin_amdgcn_wmma_f32_16x16x32_bf16(false, a.v, false, b.v,
                                                   (short)0, c, false, false);
}

// Async copy 16B from global to LDS (GV mode, per-lane addresses).
// LDS offset = low 32 bits of the generic (flat) shared-memory address.
__device__ __forceinline__ void async_b128_to_lds(const void* gptr, const void* lptr) {
    unsigned ldsoff = (unsigned)(unsigned long long)lptr;
    asm volatile("global_load_async_to_lds_b128 %0, %1, off"
                 :: "v"(ldsoff), "v"(gptr)
                 : "memory");
}

__device__ __forceinline__ void wait_async_zero() {
    asm volatile("s_wait_asynccnt 0x0" ::: "memory");
}

// ---------------------------------------------------------------------------
// Kernel 1: dinv[i] = rsqrt(1 + sum_j A[i][j])   (the +1 is the self-loop)
// ---------------------------------------------------------------------------
__global__ __launch_bounds__(256) void rowsum_dinv(const float* __restrict__ A,
                                                   float* __restrict__ dinv) {
    __shared__ float red[256];
    const int row = blockIdx.x;
    const f32x4* a4 = (const f32x4*)(A + (size_t)row * NDIM);
    float s = 0.f;
    for (int c = threadIdx.x; c < NDIM / 4; c += 256) {
        f32x4 v = a4[c];
        s += (v[0] + v[1]) + (v[2] + v[3]);
    }
    red[threadIdx.x] = s;
    __syncthreads();
    for (int off = 128; off > 0; off >>= 1) {
        if (threadIdx.x < off) red[threadIdx.x] += red[threadIdx.x + off];
        __syncthreads();
    }
    if (threadIdx.x == 0) dinv[row] = rsqrtf(red[0] + 1.0f);
}

// ---------------------------------------------------------------------------
// Kernel 2: Yt_{hi,lo}[n][j] = split_bf16( dinv[j] * X[j][n] )   (transposed)
// One block transposes a 128(j) x 128(n) tile through LDS.
// ---------------------------------------------------------------------------
#define TS 130  // padded LDS stride (ushorts)
__global__ __launch_bounds__(256) void prep_yt(const float* __restrict__ X,
                                               const float* __restrict__ dinv,
                                               unsigned short* __restrict__ Yhi,
                                               unsigned short* __restrict__ Ylo) {
    __shared__ unsigned short thi[128 * TS];
    __shared__ unsigned short tlo[128 * TS];
    const int j0 = blockIdx.x * 128;
    for (int it = 0; it < 64; ++it) {
        int lin = threadIdx.x + it * 256;          // 0..16383
        int jj = lin >> 7, n = lin & 127;
        float y = dinv[j0 + jj] * X[(size_t)(j0 + jj) * FDIM + n];
        unsigned short h, l;
        split1(y, h, l);
        thi[jj * TS + n] = h;
        tlo[jj * TS + n] = l;
    }
    __syncthreads();
    for (int it = 0; it < 64; ++it) {
        int lin = threadIdx.x + it * 256;
        int n = lin >> 7, jj = lin & 127;
        Yhi[(size_t)n * NDIM + j0 + jj] = thi[jj * TS + n];
        Ylo[(size_t)n * NDIM + j0 + jj] = tlo[jj * TS + n];
    }
}

// ---------------------------------------------------------------------------
// Kernel 3: W (fp32 [out][in]) -> bf16 hi/lo
// ---------------------------------------------------------------------------
__global__ __launch_bounds__(256) void prep_w(const float* __restrict__ W,
                                              unsigned short* __restrict__ Whi,
                                              unsigned short* __restrict__ Wlo) {
    int i = blockIdx.x * 256 + threadIdx.x;  // 16384 total
    unsigned short h, l;
    split1(W[i], h, l);
    Whi[i] = h;
    Wlo[i] = l;
}

// ---------------------------------------------------------------------------
// Kernel 4: main fused GEMM.
//   acc = A[rows] @ Y          (split-bf16, 3 WMMAs per 16x16x32 step)
//   h   = dinv_i * (acc + dinv_i * X[i,:])       (A+I diag term fused)
//   out = h @ W^T + b          (split-bf16 WMMA again, from LDS h-tile)
// Block: 128 threads (4 waves), 64 output rows, full 128 cols.
// ---------------------------------------------------------------------------
#define BM  64
#define KT  32
#define LDA 48   // LDS stride (bf16) for A tiles  -> 96B rows, 16B aligned
#define LDB 48   // LDS stride (bf16) for B tiles
#define LDH 132  // LDS stride (fp32) for h tile   -> 528B rows, 16B aligned

__global__ __launch_bounds__(128) void gcn_main(const float* __restrict__ A,
                                                const float* __restrict__ X,
                                                const float* __restrict__ bias,
                                                const float* __restrict__ dinv,
                                                const unsigned short* __restrict__ Yhi,
                                                const unsigned short* __restrict__ Ylo,
                                                const unsigned short* __restrict__ Whi,
                                                const unsigned short* __restrict__ Wlo,
                                                float* __restrict__ out) {
    __shared__ unsigned short sAhi[BM * LDA], sAlo[BM * LDA];
    __shared__ unsigned short sBhi[FDIM * LDB], sBlo[FDIM * LDB];
    __shared__ float sH[BM * LDH];

    const int tid = threadIdx.x;
    const int lane = tid & 31;
    const int wave = tid >> 5;
    const int rowBase = blockIdx.x * BM;

    const int m   = lane & 15;            // M / N position within tile
    const int kbA = (lane >> 4) << 3;     // A-frag K base: 0 or 8
    const int kbB = (lane >> 4) << 4;     // 0 or 16; unused (B frags use kbA)
    const int mhalf = (lane >> 4) << 3;   // C/D row offset: 0 or 8

    v8f acc[8] = {};

    for (int k0 = 0; k0 < NDIM; k0 += KT) {
        // ---- prefetch next A tile into cache hierarchy ----
        if (k0 + KT < NDIM) {
            __builtin_prefetch(A + (size_t)(rowBase + (tid >> 1)) * NDIM + k0 + KT, 0, 1);
        }
        // ---- stage B: Yt hi/lo chunk via async-to-LDS (ASYNCcnt path) ----
        // per array: 128*4 = 512 16B chunks, 128 threads -> 4 each
        #pragma unroll
        for (int it = 0; it < 4; ++it) {
            int idx = tid + it * 128;
            int n = idx >> 2;
            int p = (idx & 3) << 3;  // bf16 element offset: 0,8,16,24
            async_b128_to_lds(Yhi + (size_t)n * NDIM + k0 + p, sBhi + n * LDB + p);
            async_b128_to_lds(Ylo + (size_t)n * NDIM + k0 + p, sBlo + n * LDB + p);
        }
        // ---- stage A: BM x 32 fp32 -> hi/lo bf16 in LDS (non-temporal) ----
        // 64*8 = 512 float4's, 128 threads -> 4 each (coalesced)
        #pragma unroll
        for (int it = 0; it < 4; ++it) {
            int idx = tid + it * 128;
            int r = idx >> 3;
            int c4 = (idx & 7) << 2;
            f32x4 v = __builtin_nontemporal_load(
                (const f32x4*)(A + (size_t)(rowBase + r) * NDIM + k0 + c4));
            split_store4(sAhi + r * LDA + c4, sAlo + r * LDA + c4, v);
        }
        wait_async_zero();
        __syncthreads();

        // ---- A fragments (16x32): lane holds row m, K chunks kbA.. and kbA+16.. ----
        BFrag ahi, alo;
        {
            const unsigned short* pa = sAhi + (wave * 16 + m) * LDA;
            ahi.q[0] = *(const u32x4*)(pa + kbA);
            ahi.q[1] = *(const u32x4*)(pa + kbA + 16);
            const unsigned short* pl = sAlo + (wave * 16 + m) * LDA;
            alo.q[0] = *(const u32x4*)(pl + kbA);
            alo.q[1] = *(const u32x4*)(pl + kbA + 16);
        }
        // ---- 8 N-tiles, 3 WMMAs each (hi*hi + hi*lo + lo*hi) ----
        #pragma unroll
        for (int nt = 0; nt < 8; ++nt) {
            BFrag bhi, blo;
            const unsigned short* pb = sBhi + (nt * 16 + m) * LDB + kbA;
            bhi.q[0] = *(const u32x4*)(pb);
            bhi.q[1] = *(const u32x4*)(pb + 16);
            const unsigned short* pb2 = sBlo + (nt * 16 + m) * LDB + kbA;
            blo.q[0] = *(const u32x4*)(pb2);
            blo.q[1] = *(const u32x4*)(pb2 + 16);
            acc[nt] = wmma_bf16(ahi, bhi, acc[nt]);
            acc[nt] = wmma_bf16(ahi, blo, acc[nt]);
            acc[nt] = wmma_bf16(alo, bhi, acc[nt]);
        }
        __syncthreads();
    }

    // ---- epilogue 1: h = dinv_i*(acc + dinv_i*X[i,:]) -> LDS h tile ----
    #pragma unroll
    for (int v = 0; v < 8; ++v) {
        int mrow = wave * 16 + mhalf + v;
        int gi = rowBase + mrow;
        float di = dinv[gi];
        #pragma unroll
        for (int nt = 0; nt < 8; ++nt) {
            int n = nt * 16 + m;
            float hv = di * (acc[nt][v] + di * X[(size_t)gi * FDIM + n]);
            sH[mrow * LDH + n] = hv;
        }
    }
    __syncthreads();

    // ---- epilogue 2: out = h @ W^T + b  (split-bf16 WMMA from LDS) ----
    v8f acc2[8] = {};
    for (int kc = 0; kc < FDIM; kc += KT) {
        // h fragment for this wave's 16 rows
        const float* ph = sH + (wave * 16 + m) * LDH + kc;
        f32x4 f0 = *(const f32x4*)(ph + kbA);
        f32x4 f1 = *(const f32x4*)(ph + kbA + 4);
        f32x4 f2 = *(const f32x4*)(ph + kbA + 16);
        f32x4 f3 = *(const f32x4*)(ph + kbA + 20);
        BFrag hhi, hlo;
        {
            unsigned short* hp = (unsigned short*)&hhi;
            unsigned short* lp = (unsigned short*)&hlo;
            #pragma unroll
            for (int e = 0; e < 4; ++e) {
                split1(f0[e], hp[e],      lp[e]);
                split1(f1[e], hp[4 + e],  lp[4 + e]);
                split1(f2[e], hp[8 + e],  lp[8 + e]);
                split1(f3[e], hp[12 + e], lp[12 + e]);
            }
        }
        #pragma unroll
        for (int ot = 0; ot < 8; ++ot) {
            int o = ot * 16 + m;
            BFrag whi, wlo;
            const unsigned short* pw = Whi + o * FDIM + kc + kbA;
            whi.q[0] = *(const u32x4*)(pw);
            whi.q[1] = *(const u32x4*)(pw + 16);
            const unsigned short* pw2 = Wlo + o * FDIM + kc + kbA;
            wlo.q[0] = *(const u32x4*)(pw2);
            wlo.q[1] = *(const u32x4*)(pw2 + 16);
            acc2[ot] = wmma_bf16(hhi, whi, acc2[ot]);
            acc2[ot] = wmma_bf16(hhi, wlo, acc2[ot]);
            acc2[ot] = wmma_bf16(hlo, whi, acc2[ot]);
        }
    }

    // ---- store out tile + bias (C/D layout: VGPR v -> M = mhalf+v, lane%16 -> N) ----
    #pragma unroll
    for (int ot = 0; ot < 8; ++ot) {
        int col = ot * 16 + m;
        float bv = bias[col];
        #pragma unroll
        for (int v = 0; v < 8; ++v) {
            int mrow = wave * 16 + mhalf + v;
            __builtin_nontemporal_store(acc2[ot][v] + bv,
                out + (size_t)(rowBase + mrow) * FDIM + col);
        }
    }
}

// ---------------------------------------------------------------------------
// Host launcher. Inputs: X [8192,128], A [8192,8192], W [128,128], b [128].
// ---------------------------------------------------------------------------
extern "C" void kernel_launch(void* const* d_in, const int* in_sizes, int n_in,
                              void* d_out, int out_size, void* d_ws, size_t ws_size,
                              hipStream_t stream) {
    const float* X = (const float*)d_in[0];
    const float* A = (const float*)d_in[1];
    const float* W = (const float*)d_in[2];
    const float* b = (const float*)d_in[3];
    float* out = (float*)d_out;

    char* ws = (char*)d_ws;
    float* dinv = (float*)ws;                                   // 8192 * 4B
    unsigned short* Yhi = (unsigned short*)(ws + 32768);        // 128*8192 bf16
    unsigned short* Ylo = Yhi + (size_t)FDIM * NDIM;            // 128*8192 bf16
    unsigned short* Whi = Ylo + (size_t)FDIM * NDIM;            // 128*128 bf16
    unsigned short* Wlo = Whi + FDIM * FDIM;                    // 128*128 bf16

    rowsum_dinv<<<NDIM, 256, 0, stream>>>(A, dinv);
    prep_yt<<<NDIM / 128, 256, 0, stream>>>(X, dinv, Yhi, Ylo);
    prep_w<<<(FDIM * FDIM) / 256, 256, 0, stream>>>(W, Whi, Wlo);
    gcn_main<<<NDIM / BM, 128, 0, stream>>>(A, X, b, dinv, Yhi, Ylo, Whi, Wlo, out);
}
